// LocalWindowSelfAttention_51737176047942
// MI455X (gfx1250) — compile-verified, matrix operand order checked by reading, not run on hardware
//
#include <hip/hip_runtime.h>

typedef __attribute__((ext_vector_type(16))) _Float16 v16h;
typedef __attribute__((ext_vector_type(8)))  float    v8f;
typedef __attribute__((ext_vector_type(4)))  float    v4f;

#define NWIN_TOK 49
#define CDIM 256

union Frag16 { v16h h; v4f f[2]; };

// Load a 16x32 f16 A-fragment (or B-fragment: pass column-major-of-B == row-major
// of B^T) from a row-major f16 buffer with leading dimension ld (halves).
// Per CDNA5 ISA 16-bit A 16x32 layout: lanes<16 hold K {0..7,16..23}, lanes>=16
// hold K {8..15,24..31} as two contiguous 16-byte chunks.
__device__ inline v16h load_frag(const _Float16* base, int row, int ld, int k0, int lane) {
  int m  = lane & 15;
  int hi = lane >> 4;
  const _Float16* p = base + (row + m) * ld + k0 + hi * 8;
  Frag16 r;
  r.f[0] = *(const v4f*)(p);
  r.f[1] = *(const v4f*)(p + 16);
  return r.h;
}

__device__ inline v8f wmma_f16(v16h a, v16h b, v8f c) {
  return __builtin_amdgcn_wmma_f32_16x16x32_f16(false, a, false, b, (short)0, c, false, false);
}

__device__ inline v8f zero8() { v8f c; for (int r = 0; r < 8; ++r) c[r] = 0.0f; return c; }

// ---------------------------------------------------------------------------
// Kernel 0: fp32 -> f16 conversion (weights)
// ---------------------------------------------------------------------------
__global__ void cvt_f32_to_f16(const float* __restrict__ in, _Float16* __restrict__ out, int n) {
  int i = blockIdx.x * 256 + threadIdx.x;
  if (i < n) out[i] = (_Float16)in[i];
}

// ---------------------------------------------------------------------------
// Kernel 1: fused window attention. One block per (window, head).
// block = 128 threads = 4 waves.
// ---------------------------------------------------------------------------
__global__ __launch_bounds__(128) void attn_kernel(
    const float*    __restrict__ x,       // (B, 112*112, 256) fp32
    const _Float16* __restrict__ wqkv,    // (768, 256) f16
    const float*    __restrict__ qkv_b,   // (768,)
    const float*    __restrict__ rpb,     // (169, 8)
    _Float16*       __restrict__ out16)   // (200704, 256) f16, original token order
{
  __shared__ __align__(16) _Float16 xw[64 * 256];   // 32 KB window activations
  __shared__ __align__(16) _Float16 qs[64 * 32];    //  4 KB q (scaled)
  __shared__ __align__(16) _Float16 ks[64 * 32];    //  4 KB k
  __shared__ __align__(16) _Float16 vt[32 * 64];    //  4 KB v^T (dim-major)
  __shared__ __align__(16) float    S[64 * 64];     // 16 KB scores; P f16 aliases front 8KB
  _Float16* P = (_Float16*)S;

  const int blk  = blockIdx.x;
  const int head = blk & 7;
  const int win  = blk >> 3;
  const int b    = win >> 8;         // 256 windows per image
  const int nh   = (win >> 4) & 15;
  const int nw   = win & 15;
  const long grow_base = ((long)b * 112 + (long)nh * 7) * 112 + (long)nw * 7;

  const int tid  = threadIdx.x;
  const int lane = tid & 31;
  const int wid  = tid >> 5;

  // ---- stage 1: gather window rows of x -> LDS f16 (rows 49..63 zeroed) ----
  for (int c4 = tid; c4 < 4096; c4 += 128) {     // 64 rows x 64 chunks of 4 floats
    int r   = c4 >> 6;
    int col = (c4 & 63) * 4;
    v4f v; v[0] = v[1] = v[2] = v[3] = 0.0f;
    if (r < NWIN_TOK) {
      long g = grow_base + (long)(r / 7) * 112 + (r % 7);
      v = *(const v4f*)(x + g * CDIM + col);
    }
    _Float16* d = xw + r * CDIM + col;
    d[0] = (_Float16)v[0]; d[1] = (_Float16)v[1];
    d[2] = (_Float16)v[2]; d[3] = (_Float16)v[3];
  }
  __syncthreads();

  const float scale = 0.17677669529663687f;   // 32^-0.5

  // ---- stage 2: q/k/v = xw @ W^T + b for this head (M=64, N=32, K=256) ----
  // 3 mats x 4 Mtiles x 2 Ntiles = 24 WMMA tiles, 6 per wave, 8 K-steps each.
  for (int t = wid; t < 24; t += 4) {
    int mat = t >> 3;            // 0=q 1=k 2=v
    int sub = t & 7;
    int mt  = sub >> 1, nt = sub & 1;
    int wrow = mat * 256 + head * 32 + nt * 16;      // weight-row base == B-column base
    float bb = qkv_b[wrow + (lane & 15)];
    v8f c; for (int r = 0; r < 8; ++r) c[r] = bb;
    for (int k0 = 0; k0 < 256; k0 += 32) {
      v16h a  = load_frag(xw,   mt * 16, 256, k0, lane);
      v16h bf = load_frag(wqkv, wrow,    256, k0, lane);
      c = wmma_f16(a, bf, c);
    }
    int n = lane & 15, hi = lane >> 4;
    if (mat == 0) {
      for (int r = 0; r < 8; ++r)
        qs[(mt * 16 + r + hi * 8) * 32 + nt * 16 + n] = (_Float16)(c[r] * scale);
    } else if (mat == 1) {
      for (int r = 0; r < 8; ++r)
        ks[(mt * 16 + r + hi * 8) * 32 + nt * 16 + n] = (_Float16)c[r];
    } else {
      for (int r = 0; r < 8; ++r)
        vt[(nt * 16 + n) * 64 + (mt * 16 + r + hi * 8)] = (_Float16)c[r];
    }
  }
  __syncthreads();

  // ---- stage 3: S = q @ k^T  (M=64, N=64, K=32 -> single WMMA per tile) ----
  for (int t = wid; t < 16; t += 4) {
    int mt = t >> 2, nt = t & 3;
    v8f c = zero8();
    v16h a  = load_frag(qs, mt * 16, 32, 0, lane);
    v16h bf = load_frag(ks, nt * 16, 32, 0, lane);
    c = wmma_f16(a, bf, c);
    int n = lane & 15, hi = lane >> 4;
    for (int r = 0; r < 8; ++r)
      S[(mt * 16 + r + hi * 8) * 64 + nt * 16 + n] = c[r];
  }
  __syncthreads();

  // ---- stage 4: bias + masked softmax (rows 0..48), write P f16 ----
  float buf[NWIN_TOK];
  float inv = 0.0f;
  const bool active = (tid < NWIN_TOK);
  if (active) {
    int i  = tid;
    int ci = (i / 7) * 13 + (i % 7);
    float mx = -1e30f;
    for (int j = 0; j < NWIN_TOK; ++j) {
      int jr = 48 - j;
      int cj = (jr / 7) * 13 + (jr % 7);
      float val = S[i * 64 + j] + rpb[(ci + cj) * 8 + head];
      buf[j] = val;
      mx = fmaxf(mx, val);
    }
    float sum = 0.0f;
    for (int j = 0; j < NWIN_TOK; ++j) {
      float e = __expf(buf[j] - mx);
      buf[j] = e;
      sum += e;
    }
    inv = 1.0f / sum;
  }
  __syncthreads();            // all reads of S complete before aliasing writes
  if (active) {
    for (int j = 0; j < NWIN_TOK; ++j) P[tid * 64 + j] = (_Float16)(buf[j] * inv);
    for (int j = NWIN_TOK; j < 64; ++j) P[tid * 64 + j] = (_Float16)0.0f;
  } else if (tid < 64) {      // zero pad rows so fragment reads are defined
    for (int j = 0; j < 64; ++j) P[tid * 64 + j] = (_Float16)0.0f;
  }
  __syncthreads();

  // ---- stage 5: out = P @ v  (M=64, N=32, K=64) -> scatter f16 to out16 ----
  for (int t = wid; t < 8; t += 4) {
    int mt = t >> 1, nt = t & 1;
    v8f c = zero8();
    for (int k0 = 0; k0 < 64; k0 += 32) {
      v16h a  = load_frag(P,  mt * 16, 64, k0, lane);
      v16h bf = load_frag(vt, nt * 16, 64, k0, lane);  // B col n = dim -> vt row
      c = wmma_f16(a, bf, c);
    }
    int n = lane & 15, hi = lane >> 4;
    for (int r = 0; r < 8; ++r) {
      int trow = mt * 16 + r + hi * 8;
      if (trow < NWIN_TOK) {
        long g = grow_base + (long)(trow / 7) * 112 + (trow % 7);
        out16[g * CDIM + head * 32 + nt * 16 + n] = (_Float16)c[r];
      }
    }
  }
}

// ---------------------------------------------------------------------------
// Kernel 2: output projection. M=200704, N=256, K=256.
// Each wave computes a 16x64 tile (A-fragment reuse across 4 N-tiles).
// ---------------------------------------------------------------------------
__global__ __launch_bounds__(256) void proj_kernel(
    const _Float16* __restrict__ a16,     // (200704, 256) f16
    const _Float16* __restrict__ pw16,    // (256, 256) f16
    const float*    __restrict__ proj_b,  // (256,)
    float*          __restrict__ out)     // (200704, 256) fp32
{
  int gwave = (blockIdx.x * 256 + threadIdx.x) >> 5;
  int lane  = threadIdx.x & 31;
  int mtile = gwave >> 2;          // 12544 M-tiles
  int ng    = gwave & 3;           // 64-col N-group
  int n = lane & 15, hi = lane >> 4;

  v8f c[4];
  for (int q = 0; q < 4; ++q) {
    float bb = proj_b[ng * 64 + q * 16 + n];
    for (int r = 0; r < 8; ++r) c[q][r] = bb;
  }
  for (int k0 = 0; k0 < 256; k0 += 32) {
    v16h a = load_frag(a16, mtile * 16, 256, k0, lane);
    for (int q = 0; q < 4; ++q) {
      v16h bf = load_frag(pw16, ng * 64 + q * 16, 256, k0, lane);
      c[q] = wmma_f16(a, bf, c[q]);
    }
  }
  for (int q = 0; q < 4; ++q)
    for (int r = 0; r < 8; ++r)
      out[(long)(mtile * 16 + r + hi * 8) * 256 + ng * 64 + q * 16 + n] = c[q][r];
}

// ---------------------------------------------------------------------------
extern "C" void kernel_launch(void* const* d_in, const int* in_sizes, int n_in,
                              void* d_out, int out_size, void* d_ws, size_t ws_size,
                              hipStream_t stream) {
  const float* x      = (const float*)d_in[0];   // (16, 12544, 256)
  const float* qkv_w  = (const float*)d_in[1];   // (768, 256)
  const float* qkv_b  = (const float*)d_in[2];   // (768,)
  const float* proj_w = (const float*)d_in[3];   // (256, 256)
  const float* proj_b = (const float*)d_in[4];   // (256,)
  const float* rpb    = (const float*)d_in[5];   // (169, 8)
  float* out = (float*)d_out;

  char* ws = (char*)d_ws;
  _Float16* a16 = (_Float16*)ws;                              // 200704*256*2 = 102,760,448 B
  _Float16* wh  = (_Float16*)(ws + 102760448);                // 768*256*2    =     393,216 B
  _Float16* ph  = (_Float16*)(ws + 102760448 + 393216);       // 256*256*2    =     131,072 B

  cvt_f32_to_f16<<<(768 * 256 + 255) / 256, 256, 0, stream>>>(qkv_w, wh, 768 * 256);
  cvt_f32_to_f16<<<(256 * 256 + 255) / 256, 256, 0, stream>>>(proj_w, ph, 256 * 256);

  // 4096 windows * 8 heads
  attn_kernel<<<4096 * 8, 128, 0, stream>>>(x, wh, qkv_b, rpb, a16);

  // 12544 M-tiles * 4 N-groups = 50176 waves / 8 waves-per-block
  proj_kernel<<<50176 / 8, 256, 0, stream>>>(a16, ph, proj_b, out);
}
